// Transformer_30416958390636
// MI455X (gfx1250) — matrix operand order repaired and checked by reading, not compiled, not run on hardware
//
#include <hip/hip_runtime.h>
#include <hip/hip_bf16.h>
#include <math.h>

#define BB 2
#define SS 2048
#define NTOK (BB*SS)
#define DD 1024
#define HH 16
#define DKK 64
#define MFF 16
#define LL 4
#define DFFN 4096
#define VSV 69
#define VTV 69
#define FDIM 34
#define EPSF 1e-6f

typedef __attribute__((ext_vector_type(16))) __bf16 v16bf;
typedef __attribute__((ext_vector_type(8)))  float  v8f;
typedef __attribute__((ext_vector_type(4)))  unsigned int tdm_g0_t;
typedef __attribute__((ext_vector_type(8)))  int          tdm_g1_t;
typedef __attribute__((ext_vector_type(4)))  int          tdm_g2_t;
typedef __attribute__((ext_vector_type(8)))  int          tdm_g4_t;

// ---------- helpers ----------

__device__ __forceinline__ unsigned short f32_to_bf16_rne(float f) {
    unsigned u = __float_as_uint(f);
    unsigned r = u + 0x7FFFu + ((u >> 16) & 1u);
    return (unsigned short)(r >> 16);
}

__device__ __forceinline__ float gelu_f(float x) {
    float x3 = x * x * x;
    return 0.5f * x * (1.f + tanhf(0.7978845608028654f * (x + 0.044715f * x3)));
}

// 256-thread block sum (wave32: shuffle within wave, LDS across 8 waves)
__device__ __forceinline__ float block_sum256(float v, float* red) {
    #pragma unroll
    for (int o = 16; o > 0; o >>= 1) v += __shfl_xor(v, o, 32);
    __syncthreads();
    if ((threadIdx.x & 31) == 0) red[threadIdx.x >> 5] = v;
    __syncthreads();
    return red[0] + red[1] + red[2] + red[3] + red[4] + red[5] + red[6] + red[7];
}

// Issue one TDM 2D bf16 tile load: global (rows_rem x cols_rem, row stride
// stride_elems) -> LDS tile (tile_rows x tile_cols). OOB rows/cols zero-fill.
__device__ __forceinline__ void tdm_tile_load(const void* gsrc, unsigned lds_byte,
                                              int rows_rem, int cols_rem,
                                              int tile_rows, int tile_cols,
                                              int stride_elems) {
    unsigned long long ga = (unsigned long long)(uintptr_t)gsrc;
    tdm_g0_t g0;
    g0[0] = 1u;                                   // count=1, user descriptor
    g0[1] = lds_byte;                             // lds_addr (bytes)
    g0[2] = (unsigned)ga;                         // global_addr[31:0]
    g0[3] = ((unsigned)(ga >> 32) & 0x01FFFFFFu) | (2u << 30);  // addr[56:32] | type=2
    tdm_g1_t g1;
    g1[0] = 1 << 16;                              // data_size=1 (2 bytes), no mask
    g1[1] = (cols_rem & 0xFFFF) << 16;            // tensor_dim0[15:0]
    g1[2] = ((cols_rem >> 16) & 0xFFFF) | ((rows_rem & 0xFFFF) << 16); // td0 hi | td1 lo
    g1[3] = ((rows_rem >> 16) & 0xFFFF) | ((tile_cols & 0xFFFF) << 16); // td1 hi | tile_dim0
    g1[4] = tile_rows & 0xFFFF;                   // tile_dim1 (tile_dim2=0)
    g1[5] = stride_elems;                         // tensor_dim0_stride[31:0]
    g1[6] = 0;
    g1[7] = 0;
    tdm_g2_t z4 = {0, 0, 0, 0};
    tdm_g4_t z8 = {0, 0, 0, 0, 0, 0, 0, 0};
    __builtin_amdgcn_tensor_load_to_lds(g0, g1, z4, z4, z8, 0);
}

// ---------- weight conversion ----------

__global__ __launch_bounds__(256) void cvt_bf16_kernel(const float* __restrict__ s,
                                                       unsigned short* __restrict__ d, int n) {
    int i = blockIdx.x * 256 + threadIdx.x;
    if (i < n) d[i] = f32_to_bf16_rne(s[i]);
}

// ---------- embedding ----------

__global__ __launch_bounds__(256)
void embed_kernel(const int* __restrict__ ids, const float* __restrict__ id_embed,
                  const float* __restrict__ feat_tbl, const float* __restrict__ feat_proj,
                  const unsigned char* __restrict__ prod_mask, const float* __restrict__ gamma_p,
                  const float* __restrict__ pe, float* __restrict__ x) {
    __shared__ float f[FDIM];
    int token = blockIdx.x;
    int t = threadIdx.x;
    int s = token & (SS - 1);
    int id = ids[token];
    if (t < FDIM) f[t] = feat_tbl[id * FDIM + t];
    __syncthreads();
    float gm = gamma_p[0];
    float keep = prod_mask[id] ? 1.f : 0.f;
    #pragma unroll
    for (int i = 0; i < 4; ++i) {
        int d = t + 256 * i;
        float dot = 0.f;
        #pragma unroll
        for (int j = 0; j < FDIM; ++j) dot += f[j] * feat_proj[d * FDIM + j];
        x[(size_t)token * DD + d] =
            id_embed[(size_t)id * DD + d] + gm * keep * dot + pe[(size_t)s * DD + d];
    }
}

// ---------- per-user head gate softmax ----------

__global__ void gate_kernel(const int* __restrict__ user_ids,
                            const float* __restrict__ gate_logits,
                            float* __restrict__ gate) {
    int t = threadIdx.x;                 // 32 threads: lanes 0-15 = b0, 16-31 = b1
    int b = t >> 4, h = t & 15;
    float lg = gate_logits[(size_t)user_ids[b] * HH + h];
    float mx = lg;
    #pragma unroll
    for (int o = 8; o > 0; o >>= 1) mx = fmaxf(mx, __shfl_xor(mx, o, 32));
    float e = __expf(lg - mx);
    float sm = e;
    #pragma unroll
    for (int o = 8; o > 0; o >>= 1) sm += __shfl_xor(sm, o, 32);
    gate[b * HH + h] = e / sm;
}

// ---------- layernorm (torch-style: unbiased var, eps added to std) -> bf16 ----------

__global__ __launch_bounds__(256)
void ln_bf16_kernel(const float* __restrict__ x, const float* __restrict__ alpha,
                    const float* __restrict__ beta, unsigned short* __restrict__ out) {
    __shared__ float red[8];
    int row = blockIdx.x, t = threadIdx.x;
    const float* xr = x + (size_t)row * DD;
    float xv[4];
    #pragma unroll
    for (int i = 0; i < 4; ++i) xv[i] = xr[t + 256 * i];
    float s = xv[0] + xv[1] + xv[2] + xv[3];
    s = block_sum256(s, red);
    float mu = s * (1.f / (float)DD);
    float ss = 0.f;
    #pragma unroll
    for (int i = 0; i < 4; ++i) { float d0 = xv[i] - mu; ss += d0 * d0; }
    ss = block_sum256(ss, red);
    float inv = 1.f / (sqrtf(ss * (1.f / (float)(DD - 1))) + EPSF);
    #pragma unroll
    for (int i = 0; i < 4; ++i) {
        int d = t + 256 * i;
        out[(size_t)row * DD + d] = f32_to_bf16_rne(alpha[d] * (xv[i] - mu) * inv + beta[d]);
    }
}

// ---------- WMMA GEMM: C[N,NC] = act(A[N,K](bf16) @ W[NC,K]^T(bf16) + bias) [+residual] ----------
// block = 128 threads (4 waves); 64x64 C tile per block; K tiled by 64;
// A/W tiles staged by the Tensor Data Mover, double buffered, TENSORcnt-synced.

__global__ __launch_bounds__(128)
void wmma_gemm_bf16(const unsigned short* __restrict__ A, const unsigned short* __restrict__ W,
                    const float* __restrict__ bias, const float* __restrict__ residual,
                    float* __restrict__ Cf, unsigned short* __restrict__ Cb,
                    int N, int K, int NC, int act) {
    __shared__ __align__(16) unsigned short lA[2][64 * 64];
    __shared__ __align__(16) unsigned short lW[2][64 * 64];
    const int tid  = threadIdx.x;
    const int wv   = tid >> 5;
    const int lane = tid & 31;
    const int rb   = blockIdx.x * 64;
    const int cb   = blockIdx.y * 64;
    const int nk   = K >> 6;

    v8f acc[4];
    #pragma unroll
    for (int i = 0; i < 4; ++i) acc[i] = (v8f){0.f,0.f,0.f,0.f,0.f,0.f,0.f,0.f};

    if (wv == 0) {
        tdm_tile_load(&A[(size_t)rb * K], (unsigned)(uintptr_t)&lA[0][0], N - rb,  K, 64, 64, K);
        tdm_tile_load(&W[(size_t)cb * K], (unsigned)(uintptr_t)&lW[0][0], NC - cb, K, 64, 64, K);
    }

    int buf = 0;
    const int mr = lane & 15;      // row within 16 (A) / col within 16 (B)
    const int g  = lane >> 4;      // lane-group
    for (int kt = 0; kt < nk; ++kt) {
        if (wv == 0) {
            if (kt + 1 < nk) {
                int k0 = (kt + 1) << 6;
                tdm_tile_load(&A[(size_t)rb * K + k0], (unsigned)(uintptr_t)&lA[buf ^ 1][0],
                              N - rb, K - k0, 64, 64, K);
                tdm_tile_load(&W[(size_t)cb * K + k0], (unsigned)(uintptr_t)&lW[buf ^ 1][0],
                              NC - cb, K - k0, 64, 64, K);
                __builtin_amdgcn_s_wait_tensorcnt(2);   // current tile's 2 loads done
            } else {
                __builtin_amdgcn_s_wait_tensorcnt(0);
            }
        }
        __syncthreads();
        #pragma unroll
        for (int ks = 0; ks < 64; ks += 32) {
            // A fragment: halves 0..7 = K(8g..8g+7), halves 8..15 = K(16+8g..)
            union { uint4 u[2]; v16bf v; } af;
            const uint4* ap = (const uint4*)&lA[buf][((wv << 4) + mr) * 64 + ks + (g << 3)];
            af.u[0] = ap[0]; af.u[1] = ap[2];
            #pragma unroll
            for (int c4 = 0; c4 < 4; ++c4) {
                // B fragment: halves 0..7 = K(8g..8g+7), halves 8..15 = K(16+8g..) for column n=mr
                union { uint4 u[2]; v16bf v; } bfm;
                const uint4* bp = (const uint4*)&lW[buf][((c4 << 4) + mr) * 64 + ks + (g << 3)];
                bfm.u[0] = bp[0]; bfm.u[1] = bp[2];
                acc[c4] = __builtin_amdgcn_wmma_f32_16x16x32_bf16(
                    false, af.v, false, bfm.v, (short)0, acc[c4], false, false);
            }
        }
        __syncthreads();
        buf ^= 1;
    }

    // epilogue: VGPR v -> M = v + 8*g, N = lane&15
    #pragma unroll
    for (int c4 = 0; c4 < 4; ++c4) {
        #pragma unroll
        for (int v = 0; v < 8; ++v) {
            int r = rb + (wv << 4) + v + (g << 3);
            int c = cb + (c4 << 4) + mr;
            if (c < NC) {
                float val = acc[c4][v];
                if (bias)     val += bias[c];
                if (act == 1) val  = gelu_f(val);
                if (residual) val += residual[(size_t)r * NC + c];
                if (Cf) Cf[(size_t)r * NC + c] = val;
                if (Cb) Cb[(size_t)r * NC + c] = f32_to_bf16_rne(val);
            }
        }
    }
}

// ---------- performer feature map: p = exp(-0.5*(q@omega^T)^2), normalize over M ----------

__global__ __launch_bounds__(256)
void featmap_kernel(const float* __restrict__ qk, const float* __restrict__ omega,
                    float* __restrict__ out) {
    int gid   = blockIdx.x * 256 + threadIdx.x;
    int m     = gid & 15;
    int pair  = gid >> 4;          // token*H + h
    int token = pair >> 4;
    int h     = pair & 15;
    const float* qr = qk + (size_t)token * DD + h * DKK;
    const float* om = omega + m * DKK;
    float dot = 0.f;
    #pragma unroll 8
    for (int j = 0; j < DKK; ++j) dot += qr[j] * om[j];
    float ph = __expf(-0.5f * dot * dot);
    float sm = ph;
    #pragma unroll
    for (int o = 8; o > 0; o >>= 1) sm += __shfl_xor(sm, o, 32);
    out[(size_t)pair * MFF + m] = ph / (sm + EPSF);
}

// ---------- causal linear-attention scan, one block per (b,h) ----------
// state KV[M][DK] distributed: thread t owns d = t&63, m in {4*(t>>6)..+3}

__global__ __launch_bounds__(256)
void scan_kernel(const float* __restrict__ qp, const float* __restrict__ kp,
                 const float* __restrict__ v, const float* __restrict__ gate,
                 unsigned short* __restrict__ attn_b) {
    __shared__ float kps[16], qps[16], krun[16], part[256];
    __shared__ float dens;
    int b = blockIdx.x >> 4, h = blockIdx.x & 15;
    int t = threadIdx.x;
    int d = t & 63;
    int mb = t >> 6;
    float st[4] = {0.f, 0.f, 0.f, 0.f};
    if (t < 16) krun[t] = 0.f;
    float gt = gate[b * HH + h];
    for (int s = 0; s < SS; ++s) {
        int row = b * SS + s;
        size_t pidx = ((size_t)row * HH + h) * MFF;
        if (t < 16) {
            float kv = kp[pidx + t];
            kps[t] = kv;
            qps[t] = qp[pidx + t];
            krun[t] += kv;
        }
        if (s + 1 < SS) {
            __builtin_prefetch(&v[(size_t)(row + 1) * DD + h * DKK + d], 0, 0);
            __builtin_prefetch(&kp[pidx + HH * MFF + (t & 15)], 0, 0);
        }
        __syncthreads();
        float vd = v[(size_t)row * DD + h * DKK + d];
        float ns = 0.f;
        #pragma unroll
        for (int j = 0; j < 4; ++j) {
            int m = mb * 4 + j;
            st[j] += kps[m] * vd;
            ns += qps[m] * st[j];
        }
        part[t] = ns;
        if (t == 0) {
            float dd0 = 0.f;
            #pragma unroll
            for (int m = 0; m < 16; ++m) dd0 += qps[m] * krun[m];
            dens = dd0;
        }
        __syncthreads();
        if (t < 64) {
            float num = part[t] + part[t + 64] + part[t + 128] + part[t + 192];
            attn_b[(size_t)row * DD + h * DKK + t] =
                f32_to_bf16_rne(num / (dens + EPSF) * gt);
        }
        __syncthreads();
    }
}

// ---------- host orchestration ----------

extern "C" void kernel_launch(void* const* d_in, const int* in_sizes, int n_in,
                              void* d_out, int out_size, void* d_ws, size_t ws_size,
                              hipStream_t stream) {
    (void)in_sizes; (void)n_in; (void)out_size; (void)ws_size;
    const int*   ids         = (const int*)d_in[0];
    const int*   user_ids    = (const int*)d_in[1];
    const float* id_embed    = (const float*)d_in[2];
    const float* feat_proj   = (const float*)d_in[3];
    const float* gamma       = (const float*)d_in[4];
    const float* gate_logits = (const float*)d_in[5];
    const float* wq          = (const float*)d_in[6];
    const float* wk          = (const float*)d_in[7];
    const float* wv          = (const float*)d_in[8];
    const float* wo          = (const float*)d_in[9];
    const float* omega       = (const float*)d_in[10];
    const float* ln1_a       = (const float*)d_in[11];
    const float* ln1_b       = (const float*)d_in[12];
    const float* ln2_a       = (const float*)d_in[13];
    const float* ln2_b       = (const float*)d_in[14];
    const float* ff_w1       = (const float*)d_in[15];
    const float* ff_b1       = (const float*)d_in[16];
    const float* ff_w2       = (const float*)d_in[17];
    const float* ff_b2       = (const float*)d_in[18];
    const float* fin_a       = (const float*)d_in[19];
    const float* fin_b       = (const float*)d_in[20];
    const float* proj_w      = (const float*)d_in[21];
    const float* proj_b      = (const float*)d_in[22];
    const float* feat_tbl    = (const float*)d_in[23];
    const unsigned char* prod_mask = (const unsigned char*)d_in[24];
    const float* pe          = (const float*)d_in[25];

    char* ws = (char*)d_ws;
    size_t off = 0;
    auto salloc = [&](size_t bytes) -> char* {
        char* p = ws + off;
        off = (off + bytes + 255) & ~(size_t)255;
        return p;
    };
    unsigned short* wqb = (unsigned short*)salloc((size_t)LL * DD * DD * 2);
    unsigned short* wkb = (unsigned short*)salloc((size_t)LL * DD * DD * 2);
    unsigned short* wvb = (unsigned short*)salloc((size_t)LL * DD * DD * 2);
    unsigned short* wob = (unsigned short*)salloc((size_t)LL * DD * DD * 2);
    unsigned short* w1b = (unsigned short*)salloc((size_t)LL * DFFN * DD * 2);
    unsigned short* w2b = (unsigned short*)salloc((size_t)LL * DD * DFFN * 2);
    unsigned short* pwb = (unsigned short*)salloc((size_t)VTV * DD * 2);
    float* x   = (float*)salloc((size_t)NTOK * DD * 4);
    unsigned short* xnb = (unsigned short*)salloc((size_t)NTOK * DD * 2);
    float* qf  = (float*)salloc((size_t)NTOK * DD * 4);
    float* kf  = (float*)salloc((size_t)NTOK * DD * 4);
    float* vf  = (float*)salloc((size_t)NTOK * DD * 4);
    float* qpw = (float*)salloc((size_t)NTOK * HH * MFF * 4);
    float* kpw = (float*)salloc((size_t)NTOK * HH * MFF * 4);
    unsigned short* attnb = (unsigned short*)salloc((size_t)NTOK * DD * 2);
    unsigned short* hb    = (unsigned short*)salloc((size_t)NTOK * DFFN * 2);
    float* gatew = (float*)salloc((size_t)BB * HH * 4);

    auto cvt = [&](const float* s, unsigned short* dmem, int n) {
        cvt_bf16_kernel<<<(n + 255) / 256, 256, 0, stream>>>(s, dmem, n);
    };
    cvt(wq, wqb, LL * DD * DD);
    cvt(wk, wkb, LL * DD * DD);
    cvt(wv, wvb, LL * DD * DD);
    cvt(wo, wob, LL * DD * DD);
    cvt(ff_w1, w1b, LL * DFFN * DD);
    cvt(ff_w2, w2b, LL * DD * DFFN);
    cvt(proj_w, pwb, VTV * DD);

    embed_kernel<<<NTOK, 256, 0, stream>>>(ids, id_embed, feat_tbl, feat_proj,
                                           prod_mask, gamma, pe, x);
    gate_kernel<<<1, 32, 0, stream>>>(user_ids, gate_logits, gatew);

    dim3 gDD(NTOK / 64, DD / 64);
    dim3 gF1(NTOK / 64, DFFN / 64);
    for (int l = 0; l < LL; ++l) {
        ln_bf16_kernel<<<NTOK, 256, 0, stream>>>(x, ln1_a + l * DD, ln1_b + l * DD, xnb);
        wmma_gemm_bf16<<<gDD, 128, 0, stream>>>(xnb, wqb + (size_t)l * DD * DD,
            nullptr, nullptr, qf, nullptr, NTOK, DD, DD, 0);
        wmma_gemm_bf16<<<gDD, 128, 0, stream>>>(xnb, wkb + (size_t)l * DD * DD,
            nullptr, nullptr, kf, nullptr, NTOK, DD, DD, 0);
        wmma_gemm_bf16<<<gDD, 128, 0, stream>>>(xnb, wvb + (size_t)l * DD * DD,
            nullptr, nullptr, vf, nullptr, NTOK, DD, DD, 0);
        int fblocks = (NTOK * HH * MFF) / 256;
        featmap_kernel<<<fblocks, 256, 0, stream>>>(qf, omega + (size_t)l * MFF * DKK, qpw);
        featmap_kernel<<<fblocks, 256, 0, stream>>>(kf, omega + (size_t)l * MFF * DKK, kpw);
        scan_kernel<<<BB * HH, 256, 0, stream>>>(qpw, kpw, vf, gatew, attnb);
        wmma_gemm_bf16<<<gDD, 128, 0, stream>>>(attnb, wob + (size_t)l * DD * DD,
            nullptr, x, x, nullptr, NTOK, DD, DD, 0);
        ln_bf16_kernel<<<NTOK, 256, 0, stream>>>(x, ln2_a + l * DD, ln2_b + l * DD, xnb);
        wmma_gemm_bf16<<<gF1, 128, 0, stream>>>(xnb, w1b + (size_t)l * DFFN * DD,
            ff_b1 + l * DFFN, nullptr, nullptr, hb, NTOK, DD, DFFN, 1);
        wmma_gemm_bf16<<<gDD, 128, 0, stream>>>(hb, w2b + (size_t)l * DD * DFFN,
            ff_b2 + l * DD, x, x, nullptr, NTOK, DFFN, DD, 0);
    }
    ln_bf16_kernel<<<NTOK, 256, 0, stream>>>(x, fin_a, fin_b, xnb);
    dim3 gP(NTOK / 64, (VTV + 63) / 64);
    wmma_gemm_bf16<<<gP, 128, 0, stream>>>(xnb, pwb, proj_b, nullptr,
        (float*)d_out, nullptr, NTOK, DD, VTV, 0);
}
